// Transformer_87703232184452
// MI455X (gfx1250) — matrix operand order repaired and checked by reading, not compiled, not run on hardware
//
#include <hip/hip_runtime.h>
#include <hip/hip_bf16.h>

// Transformer forward (faithful to the reference, including the masked-raw-score
// "bug": attn = (scale*QK^T - 1e9*mask) @ V, NO softmax).
// All GEMM-shaped math: v_wmma_f32_16x16x32_bf16, fp32 accumulate.
// Activations feeding GEMM A-operands keep a bf16 shadow in global memory so
// A-tile / K-tile LDS staging uses GLOBAL_LOAD_ASYNC_TO_LDS_B128 (ASYNCcnt).

#define L_  6
#define NH_ 16
#define H_  1024
#define DH_ 64
#define S_  2048
#define V_  50257
#define FF_ 4096

typedef __attribute__((ext_vector_type(16))) __bf16 v16bf;
typedef __attribute__((ext_vector_type(8)))  float  v8f;

union AFrag { v16bf v; unsigned int u[8]; unsigned short h[16]; };
union CFrag { v8f   v; float f[8]; };

__device__ __forceinline__ unsigned short f2bf(float f) {
    union { float f; unsigned int u; } x; x.f = f;
    unsigned int u = x.u;
    u += 0x7FFFu + ((u >> 16) & 1u);           // RNE
    return (unsigned short)(u >> 16);
}
__device__ __forceinline__ unsigned int f2bf_pk(float a, float b) {
    return (unsigned)f2bf(a) | ((unsigned)f2bf(b) << 16);
}

// A-matrix 16x32 bf16 fragment K mapping (ISA 7.12.2): VGPR j, lane-half h
__device__ __forceinline__ int kmapA(int j, int half) {
    return ((j < 4) ? 2 * j : 2 * j + 8) + 8 * half;
}
// B-matrix 32x16 bf16 fragment K mapping (same as A): VGPR j, lane-half h
__device__ __forceinline__ int kmapB(int j, int half) {
    return ((j < 4) ? 2 * j : 2 * j + 8) + 8 * half;
}

// ---- CDNA5 async global->LDS copy (ASYNCcnt) ------------------------------
__device__ __forceinline__ unsigned lds_off32(const void* p) {
    return (unsigned)(unsigned long long)p;    // LDS generic addr low 32 bits
}
__device__ __forceinline__ void async_copy_b128(unsigned lds_addr, const void* g) {
    asm volatile("global_load_async_to_lds_b128 %0, %1, off"
                 :: "v"(lds_addr), "v"((unsigned long long)g) : "memory");
}
__device__ __forceinline__ void wait_async0() {
    asm volatile("s_wait_asynccnt 0" ::: "memory");
}

// ---------------------------------------------------------------------------
// GEMM: C[M,N] = Abf16[M,K] @ Bfp32[K,N] (+bias) -> fp32 and/or bf16 output.
// 256 threads = 8 waves as 4x2, wave tile 32x32 (2x2 frags), block 128x64xBK32.
// ---------------------------------------------------------------------------
#define BM 128
#define BN 64
#define BK 32
#define LDAS 40   // elems; 80B row stride (16B multiple for async b128 writes)
#define LDBS 34   // elems; BsT[n][k] VALU-staged, u32 aligned

__global__ __launch_bounds__(256)
void gemm_bf16_kernel(const unsigned short* __restrict__ A, const float* __restrict__ B,
                      const float* __restrict__ bias,
                      float* __restrict__ Cf, unsigned short* __restrict__ Cbf,
                      int M, int N, int K, long long sB, long long sC)
{
    __shared__ unsigned short As[BM * LDAS];   // 10240 B
    __shared__ unsigned short BsT[BN * LDBS];  //  4352 B

    const int bz = blockIdx.z;
    B += (long long)bz * sB;

    const int bm0 = blockIdx.x * BM;
    const int bn0 = blockIdx.y * BN;
    const int tid  = threadIdx.x;
    const int lane = tid & 31;
    const int wave = tid >> 5;
    const int wm = (wave >> 1) * 32;           // 0,32,64,96
    const int wn = (wave & 1) * 32;            // 0,32
    const int half = lane >> 4;
    const int l15  = lane & 15;

    const unsigned asBase = lds_off32(&As[0]);
    const bool fastB = ((N & 3) == 0) && (bn0 + BN <= N);

    // A staging addresses (bf16 raw async copy): thread -> row, 16-elem segment
    const int arow = tid >> 1;                 // 0..127
    const int aseg = (tid & 1) * 16;           // 0 or 16
    const unsigned aDst = asBase + (unsigned)(arow * LDAS + aseg) * 2;
    const unsigned short* aSrc = A + (long long)(bm0 + arow) * K + aseg;

    // B staging: thread -> 2 k rows x 4 n cols
    const int bkk = (tid >> 4) * 2;            // 0..30
    const int bn4 = (tid & 15) * 4;            // 0..60

    CFrag c[2][2];
#pragma unroll
    for (int i = 0; i < 2; i++)
#pragma unroll
        for (int j = 0; j < 2; j++)
#pragma unroll
            for (int r = 0; r < 8; r++) c[i][j].f[r] = 0.f;

    for (int k0 = 0; k0 < K; k0 += BK) {
        __syncthreads();
        // ---- async copy A tile [128 x 32] bf16 ----
        async_copy_b128(aDst,      aSrc + k0);
        async_copy_b128(aDst + 16, aSrc + k0 + 8);

        // ---- stage B tile [32 x 64] fp32 -> bf16 transposed [n][k] ----
        {
            const float* src0 = B + (long long)(k0 + bkk) * N + bn0 + bn4;
            const float* src1 = src0 + N;
            if (fastB) {
                float4 r0 = *(const float4*)src0;
                float4 r1 = *(const float4*)src1;
                if (k0 + BK < K)
                    __builtin_prefetch((const void*)(src0 + (long long)BK * N), 0, 1);
                *(unsigned int*)&BsT[(bn4 + 0) * LDBS + bkk] = f2bf_pk(r0.x, r1.x);
                *(unsigned int*)&BsT[(bn4 + 1) * LDBS + bkk] = f2bf_pk(r0.y, r1.y);
                *(unsigned int*)&BsT[(bn4 + 2) * LDBS + bkk] = f2bf_pk(r0.z, r1.z);
                *(unsigned int*)&BsT[(bn4 + 3) * LDBS + bkk] = f2bf_pk(r0.w, r1.w);
            } else {
#pragma unroll
                for (int i = 0; i < 4; i++) {
                    const int gn = bn0 + bn4 + i;
                    float f0 = (gn < N) ? src0[i] : 0.f;
                    float f1 = (gn < N) ? src1[i] : 0.f;
                    *(unsigned int*)&BsT[(bn4 + i) * LDBS + bkk] = f2bf_pk(f0, f1);
                }
            }
        }
        wait_async0();
        __syncthreads();

        AFrag a0, a1, b0, b1;
#pragma unroll
        for (int j = 0; j < 8; j++) {
            const int ka = kmapA(j, half);
            const int kb = kmapB(j, half);
            a0.u[j] = *(const unsigned int*)&As[(wm + l15) * LDAS + ka];
            a1.u[j] = *(const unsigned int*)&As[(wm + 16 + l15) * LDAS + ka];
            b0.u[j] = *(const unsigned int*)&BsT[(wn + l15) * LDBS + kb];
            b1.u[j] = *(const unsigned int*)&BsT[(wn + 16 + l15) * LDBS + kb];
        }
        c[0][0].v = __builtin_amdgcn_wmma_f32_16x16x32_bf16(false, a0.v, false, b0.v, (short)0, c[0][0].v, false, false);
        c[0][1].v = __builtin_amdgcn_wmma_f32_16x16x32_bf16(false, a0.v, false, b1.v, (short)0, c[0][1].v, false, false);
        c[1][0].v = __builtin_amdgcn_wmma_f32_16x16x32_bf16(false, a1.v, false, b0.v, (short)0, c[1][0].v, false, false);
        c[1][1].v = __builtin_amdgcn_wmma_f32_16x16x32_bf16(false, a1.v, false, b1.v, (short)0, c[1][1].v, false, false);
    }

#pragma unroll
    for (int j = 0; j < 2; j++) {
        const int col = bn0 + wn + j * 16 + l15;
        if (col >= N) continue;
        const float bv = bias ? bias[col] : 0.f;
#pragma unroll
        for (int i = 0; i < 2; i++) {
            const int row0 = bm0 + wm + i * 16;
#pragma unroll
            for (int r = 0; r < 8; r++) {
                const int m = r + 8 * half;
                const float val = c[i][j].f[r] + bv;
                const long long idx = (long long)(row0 + m) * N + col + (long long)bz * sC;
                if (Cf)  Cf[idx]  = val;
                if (Cbf) Cbf[idx] = f2bf(val);
            }
        }
    }
}

// ---------------------------------------------------------------------------
// Attention: att = (scale*QK^T - 1e9*(t>s)) @ V, full SxS, bf16 in, bf16 out.
// grid = (S/128, NH); 8 waves, each owns 16 query rows.
// ---------------------------------------------------------------------------
#define LDKB 72   // Kb row stride (elems); 144B = 16B multiple for async b128

__global__ __launch_bounds__(256)
void attention_kernel(const unsigned short* __restrict__ q,
                      const unsigned short* __restrict__ k,
                      const unsigned short* __restrict__ v,
                      unsigned short* __restrict__ att)
{
    __shared__ unsigned short Kb[32 * LDKB];      // [key][dh]   4608 B
    __shared__ unsigned short VT[64 * 34];        // [dh][key]   4352 B
    __shared__ unsigned short Sb[8 * 16 * 34];    // per-wave scores 8704 B

    const int head = blockIdx.y;
    const int qblk = blockIdx.x * 128;
    const int tid  = threadIdx.x;
    const int lane = tid & 31;
    const int wave = tid >> 5;
    const int half = lane >> 4;
    const int l15  = lane & 15;
    const float scale = 0.125f;                   // 1/sqrt(DH)

    const unsigned short* qh = q + (long long)head * S_ * DH_;
    const unsigned short* kh = k + (long long)head * S_ * DH_;
    const unsigned short* vh = v + (long long)head * S_ * DH_;

    // resident Q fragments for dh [0,32) and [32,64)
    AFrag qa[2];
    const int qrow = qblk + wave * 16 + l15;
#pragma unroll
    for (int dh = 0; dh < 2; dh++)
#pragma unroll
        for (int j = 0; j < 8; j++) {
            const int kk = kmapA(j, half) + dh * 32;
            qa[dh].u[j] = *(const unsigned int*)&qh[(long long)qrow * DH_ + kk];
        }

    CFrag acc[4];
#pragma unroll
    for (int d = 0; d < 4; d++)
#pragma unroll
        for (int r = 0; r < 8; r++) acc[d].f[r] = 0.f;

    // staging maps: 32 keys x 64 dh; one async b128 per thread for K
    const int srow = tid >> 3;                    // key 0..31
    const int sseg = (tid & 7) * 8;               // dh 0..56
    const unsigned kDst = lds_off32(&Kb[0]) + (unsigned)(srow * LDKB + sseg) * 2;

    for (int kt = 0; kt < S_; kt += 32) {
        __syncthreads();
        async_copy_b128(kDst, kh + (long long)(kt + srow) * DH_ + sseg);
        // V^T staging (needs transpose -> VALU/ds path)
        {
            const unsigned short* vs = vh + (long long)(kt + srow) * DH_ + sseg;
#pragma unroll
            for (int i = 0; i < 4; i++) {
                const unsigned int p = *(const unsigned int*)&vs[2 * i];
                VT[(sseg + 2 * i) * 34 + srow]     = (unsigned short)(p & 0xFFFF);
                VT[(sseg + 2 * i + 1) * 34 + srow] = (unsigned short)(p >> 16);
            }
        }
        wait_async0();
        __syncthreads();

        // scores = scale * q @ k^T - 1e9*(t>s), two 16-key sub-tiles
#pragma unroll
        for (int tsub = 0; tsub < 2; tsub++) {
            CFrag sc;
#pragma unroll
            for (int r = 0; r < 8; r++) sc.f[r] = 0.f;
            const int keyrow = tsub * 16 + l15;
#pragma unroll
            for (int dh = 0; dh < 2; dh++) {
                AFrag kb;
#pragma unroll
                for (int j = 0; j < 8; j++)
                    kb.u[j] = *(const unsigned int*)
                              &Kb[keyrow * LDKB + dh * 32 + kmapB(j, half)];
                sc.v = __builtin_amdgcn_wmma_f32_16x16x32_bf16(
                           false, qa[dh].v, false, kb.v, (short)0, sc.v, false, false);
            }
            const int tcol = kt + tsub * 16 + l15;
#pragma unroll
            for (int r = 0; r < 8; r++) {
                const int m = r + 8 * half;
                const int sg = qblk + wave * 16 + m;
                float val = sc.f[r] * scale - ((tcol > sg) ? 1.0e9f : 0.f);
                Sb[wave * (16 * 34) + m * 34 + tsub * 16 + l15] = f2bf(val);
            }
        }

        // acc += scores[16x32] @ V[32x64]
        AFrag sa;
#pragma unroll
        for (int j = 0; j < 8; j++)
            sa.u[j] = *(const unsigned int*)
                      &Sb[wave * (16 * 34) + l15 * 34 + kmapA(j, half)];
#pragma unroll
        for (int d = 0; d < 4; d++) {
            AFrag vb;
#pragma unroll
            for (int j = 0; j < 8; j++)
                vb.u[j] = *(const unsigned int*)
                          &VT[(d * 16 + l15) * 34 + kmapB(j, half)];
            acc[d].v = __builtin_amdgcn_wmma_f32_16x16x32_bf16(
                           false, sa.v, false, vb.v, (short)0, acc[d].v, false, false);
        }
    }

    // concat-head output, bf16 (feeds proj GEMM A operand)
#pragma unroll
    for (int d = 0; d < 4; d++) {
        const int col = head * DH_ + d * 16 + l15;
#pragma unroll
        for (int r = 0; r < 8; r++) {
            const int m = r + 8 * half;
            att[(long long)(qblk + wave * 16 + m) * H_ + col] = f2bf(acc[d].f[r]);
        }
    }
}

// ---------------------------------------------------------------------------
// Elementwise / LayerNorm kernels
// ---------------------------------------------------------------------------
__global__ __launch_bounds__(256)
void embed_kernel(const int* __restrict__ tokens, const float* __restrict__ emb,
                  const float* __restrict__ pos,
                  float* __restrict__ xf, unsigned short* __restrict__ xbf)
{
    const int idx = blockIdx.x * 256 + threadIdx.x;   // over S*H
    const int s = idx >> 10;
    const int h = idx & (H_ - 1);
    const float v = emb[(long long)tokens[s] * H_ + h] + pos[idx];
    xf[idx]  = v;
    xbf[idx] = f2bf(v);
}

__global__ __launch_bounds__(256)
void gelu_kernel(const float* __restrict__ x, unsigned short* __restrict__ obf)
{
    const int idx = blockIdx.x * 256 + threadIdx.x;
    const float v = x[idx];
    obf[idx] = f2bf(0.5f * v * (1.f + erff(v * 0.70710678118654752f)));
}

// outf/outbf = (resid ? resid : 0) + LayerNorm(src) ; one block per row
__global__ __launch_bounds__(256)
void ln_residual_kernel(const float* __restrict__ src, const float* __restrict__ resid,
                        const float* __restrict__ g, const float* __restrict__ b,
                        float* __restrict__ outf, unsigned short* __restrict__ outbf)
{
    __shared__ float rs[256], rs2[256];
    const int row = blockIdx.x;
    const float* x = src + (long long)row * H_;
    float vals[4];
    float s = 0.f, s2 = 0.f;
#pragma unroll
    for (int i = 0; i < 4; i++) {
        const float vv = x[threadIdx.x + i * 256];
        vals[i] = vv; s += vv; s2 += vv * vv;
    }
    rs[threadIdx.x] = s; rs2[threadIdx.x] = s2;
    __syncthreads();
    for (int off = 128; off > 0; off >>= 1) {
        if (threadIdx.x < off) {
            rs[threadIdx.x]  += rs[threadIdx.x + off];
            rs2[threadIdx.x] += rs2[threadIdx.x + off];
        }
        __syncthreads();
    }
    const float mean = rs[0] * (1.f / H_);
    const float var  = rs2[0] * (1.f / H_) - mean * mean;
    const float inv  = rsqrtf(var + 1e-5f);
#pragma unroll
    for (int i = 0; i < 4; i++) {
        const int cidx = threadIdx.x + i * 256;
        float o = (vals[i] - mean) * inv * g[cidx] + b[cidx];
        if (resid) o += resid[(long long)row * H_ + cidx];
        const long long oi = (long long)row * H_ + cidx;
        if (outf)  outf[oi]  = o;
        if (outbf) outbf[oi] = f2bf(o);
    }
}

// ---------------------------------------------------------------------------
static inline void launch_gemm(const unsigned short* A, const float* B, const float* bias,
                               float* Cf, unsigned short* Cbf, int M, int N, int K,
                               long long sB, long long sC, int batch, hipStream_t st)
{
    dim3 grid((M + BM - 1) / BM, (N + BN - 1) / BN, batch);
    gemm_bf16_kernel<<<grid, 256, 0, st>>>(A, B, bias, Cf, Cbf, M, N, K, sB, sC);
}

extern "C" void kernel_launch(void* const* d_in, const int* in_sizes, int n_in,
                              void* d_out, int out_size, void* d_ws, size_t ws_size,
                              hipStream_t stream)
{
    const int*   tokens = (const int*)  d_in[0];
    const float* emb    = (const float*)d_in[1];
    const float* pos    = (const float*)d_in[2];
    const float* Wq     = (const float*)d_in[3];
    const float* Wk     = (const float*)d_in[4];
    const float* Wv     = (const float*)d_in[5];
    const float* Wp     = (const float*)d_in[6];
    const float* bp     = (const float*)d_in[7];
    const float* W1     = (const float*)d_in[8];
    const float* b1     = (const float*)d_in[9];
    const float* W2     = (const float*)d_in[10];
    const float* b2     = (const float*)d_in[11];
    const float* ln_g   = (const float*)d_in[12];
    const float* ln_b   = (const float*)d_in[13];
    const float* Wlm    = (const float*)d_in[14];
    const float* blm    = (const float*)d_in[15];
    float* out = (float*)d_out;

    const long long SH = (long long)S_ * H_;      // 2,097,152
    // fp32 region: x, xa, proj, mlp (1 SH each) + h1 (4 SH)  = 8*SH floats
    float* x    = (float*)d_ws;
    float* xa   = x    + SH;
    float* proj = xa   + SH;
    float* mlp  = proj + SH;
    float* h1   = mlp  + SH;                       // 4*SH
    // bf16 region: 11*SH ushorts (total ws ~ 113 MB)
    unsigned short* xbf   = (unsigned short*)(h1 + 4 * SH);
    unsigned short* xabf  = xbf   + SH;
    unsigned short* attbf = xabf  + SH;
    unsigned short* xlnbf = attbf + SH;
    unsigned short* qbf   = xlnbf + SH;
    unsigned short* kbf   = qbf   + SH;
    unsigned short* vbf   = kbf   + SH;
    unsigned short* h1bf  = vbf   + SH;            // 4*SH

    embed_kernel<<<(int)(SH / 256), 256, 0, stream>>>(tokens, emb, pos, x, xbf);

    for (int l = 0; l < L_; l++) {
        const long long wqkv_l = (long long)l * NH_ * H_ * DH_;
        // per-head projections: [S,H] @ [H,DH] -> bf16 [NH,S,DH]
        launch_gemm(xbf, Wq + wqkv_l, nullptr, nullptr, qbf, S_, DH_, H_,
                    (long long)H_ * DH_, (long long)S_ * DH_, NH_, stream);
        launch_gemm(xbf, Wk + wqkv_l, nullptr, nullptr, kbf, S_, DH_, H_,
                    (long long)H_ * DH_, (long long)S_ * DH_, NH_, stream);
        launch_gemm(xbf, Wv + wqkv_l, nullptr, nullptr, vbf, S_, DH_, H_,
                    (long long)H_ * DH_, (long long)S_ * DH_, NH_, stream);

        attention_kernel<<<dim3(S_ / 128, NH_), 256, 0, stream>>>(qbf, kbf, vbf, attbf);

        // proj: att @ Wp[l] + bp[l] -> fp32
        launch_gemm(attbf, Wp + (long long)l * H_ * H_, bp + (long long)l * H_,
                    proj, nullptr, S_, H_, H_, 0, 0, 1, stream);

        // xa = x + LN(proj)   (fp32 + bf16 shadow)
        ln_residual_kernel<<<S_, 256, 0, stream>>>(proj, x, ln_g, ln_b, xa, xabf);

        // h1 = xa @ W1[l] + b1[l] ; gelu -> bf16
        launch_gemm(xabf, W1 + (long long)l * H_ * FF_, b1 + (long long)l * FF_,
                    h1, nullptr, S_, FF_, H_, 0, 0, 1, stream);
        gelu_kernel<<<(int)((long long)S_ * FF_ / 256), 256, 0, stream>>>(h1, h1bf);

        // mlp = gelu(h1) @ W2[l] + b2[l]
        launch_gemm(h1bf, W2 + (long long)l * FF_ * H_, b2 + (long long)l * H_,
                    mlp, nullptr, S_, H_, FF_, 0, 0, 1, stream);

        // x = xa + LN(mlp)
        ln_residual_kernel<<<S_, 256, 0, stream>>>(mlp, xa, ln_g, ln_b, x, xbf);
    }

    // final LN (bf16 only) then LM head: out = LN(x) @ Wlm + blm  [S, V] fp32
    ln_residual_kernel<<<S_, 256, 0, stream>>>(x, nullptr, ln_g, ln_b, nullptr, xlnbf);
    launch_gemm(xlnbf, Wlm, blm, out, nullptr, S_, V_, H_, 0, 0, 1, stream);
}
